// CrossAttentionModel_39101382263139
// MI455X (gfx1250) — matrix operand order repaired and checked by reading, not compiled, not run on hardware
//
#include <hip/hip_runtime.h>
#include <hip/hip_bf16.h>

// ---- problem constants ----
#define B_  8
#define L_  64
#define D_  768
#define H_  1024
#define PADID 50257

typedef __bf16 bf16_t;
typedef bf16_t v16bf __attribute__((ext_vector_type(16)));
typedef float  v8f   __attribute__((ext_vector_type(8)));

__device__ __forceinline__ v8f wmma_bf16(v16bf a, v16bf b, v8f c) {
  // D = A(16x32 bf16) * B(32x16 bf16) + C(16x16 f32)
  return __builtin_amdgcn_wmma_f32_16x16x32_bf16(false, a, false, b, (short)0, c,
                                                 false, false);
}

// CDNA5 async global->LDS copy (ASYNCcnt-tracked); VDST VGPR = LDS byte offset.
__device__ __forceinline__ void async_ld16(unsigned int lds_byte_off,
                                           unsigned long long gaddr) {
  asm volatile("global_load_async_to_lds_b128 %0, %1, off"
               :
               : "v"(lds_byte_off), "v"(gaddr)
               : "memory");
}
__device__ __forceinline__ void wait_async0() {
  asm volatile("s_wait_asynccnt 0x0" ::: "memory");
}

// A-fragment packed layout for 16-bit A (16x32 per block), per ISA 7.12.2:
//   lane = half*16 + (M%16); per-lane 16 contiguous bf16: j -> K mapping
__device__ __forceinline__ int apack_k_of_j(int j, int half) {
  int v = j >> 1, pos = j & 1;
  return (v < 4) ? (v * 2 + pos + half * 8) : (16 + (v - 4) * 2 + pos + half * 8);
}
// inverse: given column kk (0..31) inside a K-block -> (half, j)
__device__ __forceinline__ void apack_inv(int kk, int& half, int& j) {
  half = (kk >> 3) & 1;
  j = ((kk >> 4) << 3) | (kk & 7);
}

// ---------------------------------------------------------------------------
// Kernel 1a: gather embedding rows into bf16 A-fragment layout (PAD -> 0)
// e1A/e2A: [MB=32][KB=24][lane 32][j 16]  (M = B*L = 512 rows, K = D = 768)
// ---------------------------------------------------------------------------
__global__ void pack_embed_kernel(const int* __restrict__ x1,
                                  const int* __restrict__ x2,
                                  const float* __restrict__ embed,
                                  bf16_t* __restrict__ e1A,
                                  bf16_t* __restrict__ e2A) {
  const int N1 = 512 * 768;
  int idx = blockIdx.x * blockDim.x + threadIdx.x;
  if (idx >= 2 * N1) return;
  int which = idx / N1;
  int t = idx - which * N1;
  int j = t & 15, lane = (t >> 4) & 31, blk = t >> 9;
  int kb = blk % 24, mb = blk / 24;
  int l = mb * 16 + (lane & 15);
  int half = lane >> 4;
  int c = kb * 32 + apack_k_of_j(j, half);
  int tok = which == 0 ? x1[l] : x2[l];
  if (tok == PADID) tok = 0;
  float val = embed[(size_t)tok * D_ + c];
  (which == 0 ? e1A : e2A)[t] = (bf16_t)val;
}

// ---------------------------------------------------------------------------
// Kernel 1b: pack a row-major f32 [K x N] weight into bf16 B-fragment layout.
// B-frag per ISA: lane L holds N=L%16, K = apack_k_of_j(j, L/16) (same order as A)
// offset(kb,nb,lane,j) = ((kb*NB+nb)*32+lane)*16 + j
// ---------------------------------------------------------------------------
__global__ void pack_B_kernel(const float* __restrict__ src,
                              bf16_t* __restrict__ dst, int K, int N) {
  int NB = N >> 4;
  int total = K * N;
  int t = blockIdx.x * blockDim.x + threadIdx.x;
  if (t >= total) return;
  int j = t & 15, lane = (t >> 4) & 31, blk = t >> 9;
  int nb = blk % NB, kb = blk / NB;
  int kk = kb * 32 + apack_k_of_j(j, lane >> 4);
  int nn = nb * 16 + (lane & 15);
  dst[t] = (bf16_t)src[(size_t)kk * N + nn];
}

// ---------------------------------------------------------------------------
// Kernel 2: h1 = e1 @ w1a, h2 = e2 @ w1b   (M=512, K=768, N=1024, f32 out)
// Grid: 256 blocks (which[2] x mblk[8] x nblk[16]), 256 thr = 8 waves.
// ---------------------------------------------------------------------------
__global__ void __launch_bounds__(256) gemm_h_kernel(
    const bf16_t* __restrict__ e1A, const bf16_t* __restrict__ e2A,
    const bf16_t* __restrict__ w1aP, const bf16_t* __restrict__ w1bP,
    float* __restrict__ h1f, float* __restrict__ h2f) {
  int bid = blockIdx.x;
  int which = bid >> 7;
  int rem = bid & 127;
  int mblk = rem >> 4, nblk = rem & 15;
  const bf16_t* A = which ? e2A : e1A;
  const bf16_t* W = which ? w1bP : w1aP;
  float* out = which ? h2f : h1f;
  int w = threadIdx.x >> 5, lane = threadIdx.x & 31;
  int gmt = mblk * 4 + (w & 3);
  int gnt0 = nblk * 4 + (w >> 2) * 2;
  v8f c0 = {0.f, 0.f, 0.f, 0.f, 0.f, 0.f, 0.f, 0.f};
  v8f c1 = {0.f, 0.f, 0.f, 0.f, 0.f, 0.f, 0.f, 0.f};
  for (int kb = 0; kb < 24; ++kb) {
    const bf16_t* ap = A + ((gmt * 24 + kb) * 32 + lane) * 16;
    const bf16_t* bp0 = W + ((kb * 64 + gnt0) * 32 + lane) * 16;
    const bf16_t* bp1 = W + ((kb * 64 + gnt0 + 1) * 32 + lane) * 16;
    if (kb + 1 < 24) {
      __builtin_prefetch(A + ((gmt * 24 + kb + 1) * 32 + lane) * 16, 0, 1);
      __builtin_prefetch(W + (((kb + 1) * 64 + gnt0) * 32 + lane) * 16, 0, 1);
    }
    v16bf a = *(const v16bf*)ap;
    v16bf b0 = *(const v16bf*)bp0;
    v16bf b1 = *(const v16bf*)bp1;
    c0 = wmma_bf16(a, b0, c0);
    c1 = wmma_bf16(a, b1, c1);
  }
  int half = lane >> 4, nlo = lane & 15;
#pragma unroll
  for (int r = 0; r < 8; ++r) {
    int row = gmt * 16 + half * 8 + r;
    out[(size_t)row * H_ + gnt0 * 16 + nlo] = c0[r];
    out[(size_t)row * H_ + (gnt0 + 1) * 16 + nlo] = c1[r];
  }
}

// ---------------------------------------------------------------------------
// Kernel 3: zero the pooled-numerator / denominator accumulators
// ---------------------------------------------------------------------------
__global__ void zero_kernel(float* __restrict__ p, int n) {
  int t = blockIdx.x * blockDim.x + threadIdx.x;
  if (t < n) p[t] = 0.f;
}

// ---------------------------------------------------------------------------
// Kernel 4: fused per-(b,m) pipeline. 512 WGs x 256 threads.
// LDS map (229888 B dynamic):
//   [0      .. 131071] hB   (64x1024 bf16, A-frag)  -- reused in phase 3 as the
//                            async double-buffer for aw1 B-frags (2 x 48 KB)
//   [131072 .. 229375] embB (64x768 bf16, A-frag)
//   [229376 .. 229631] logitsBuf (64 f32)
//   [229632 .. 229887] wBuf (64 f32)
// ---------------------------------------------------------------------------
__global__ void __launch_bounds__(256) cross_main_kernel(
    const int* __restrict__ x1, const int* __restrict__ x2,
    const int* __restrict__ mask1, const int* __restrict__ mask2,
    const float* __restrict__ h1f, const float* __restrict__ h2f,
    const float* __restrict__ tb1, const bf16_t* __restrict__ tw2P,
    const float* __restrict__ tb2, const bf16_t* __restrict__ aw1P,
    const float* __restrict__ ab1, const float* __restrict__ aw2,
    const float* __restrict__ ab2, float* __restrict__ pool,
    float* __restrict__ den) {
  extern __shared__ char lds[];
  bf16_t* hB = (bf16_t*)lds;
  bf16_t* embB = (bf16_t*)(lds + 131072);
  float* logitsBuf = (float*)(lds + 229376);
  float* wBuf = (float*)(lds + 229632);

  int bb = blockIdx.x >> 6;
  int mm = blockIdx.x & 63;
  int tid = threadIdx.x;
  int w = tid >> 5, lane = tid & 31;

  if (tid < 64) logitsBuf[tid] = ab2[0];

  // Phase 1: hB = bf16(relu(h1[b,l,:] + h2[b,m,:] + tb1)) in A-frag layout
  const float* h2row = h2f + (size_t)(bb * 64 + mm) * H_;
  for (int idx = tid; idx < 64 * H_; idx += 256) {
    int j = idx & 15, ln = (idx >> 4) & 31, blk = idx >> 9;
    int kb = blk & 31, mb = blk >> 5;
    int l = mb * 16 + (ln & 15);
    int half = ln >> 4;
    int hc = kb * 32 + apack_k_of_j(j, half);
    float val = h1f[(size_t)(bb * 64 + l) * H_ + hc] + h2row[hc] + tb1[hc];
    hB[idx] = (bf16_t)fmaxf(val, 0.f);
  }
  __syncthreads();

  // Phase 2: emb = hB @ tw2 + tb2  (M=64,K=1024,N=768) -> embB (A-frag, bf16)
  int mt = w & 3;
  int parity = w >> 2;  // waves 0-3: even ntiles, waves 4-7: odd
  for (int i = 0; i < 24; ++i) {
    int nt = parity + i * 2;
    v8f c = {0.f, 0.f, 0.f, 0.f, 0.f, 0.f, 0.f, 0.f};
    for (int kb = 0; kb < 32; ++kb) {
      v16bf a = *(const v16bf*)(hB + ((mt * 32 + kb) * 32 + lane) * 16);
      if (kb + 1 < 32)
        __builtin_prefetch(tw2P + (((size_t)(kb + 1) * 48 + nt) * 32 + lane) * 16, 0, 1);
      v16bf bfr = *(const v16bf*)(tw2P + (((size_t)kb * 48 + nt) * 32 + lane) * 16);
      c = wmma_bf16(a, bfr, c);
    }
    int col = nt * 16 + (lane & 15);
    float bias = tb2[col];
    int rowhalf = lane >> 4;
    int kb2 = col >> 5, kk = col & 31, khalf, jj;
    apack_inv(kk, khalf, jj);
#pragma unroll
    for (int r = 0; r < 8; ++r) {
      int l = mt * 16 + rowhalf * 8 + r;
      int ln2 = (khalf << 4) | (l & 15);
      int mb2 = l >> 4;
      embB[((mb2 * 24 + kb2) * 32 + ln2) * 16 + jj] = (bf16_t)(c[r] + bias);
    }
  }
  __syncthreads();  // hB is dead from here on -> reuse as async staging buffer

  // Phase 3: a = relu(embB @ aw1 + ab1); logits[l] += sum_col a*aw2 (fused).
  // aw1 B-fragments are staged into LDS once per WG with async global->LDS
  // copies (double-buffered 2 x 48 KB in the dead hB region), so all 8 waves
  // share one copy instead of 8 separate L2 streams.
  const char* aw1Pc = (const char*)aw1P;
  unsigned int stageBase = (unsigned int)(uintptr_t)lds;  // low 32 bits = LDS off
  auto issue_stage = [&](int buf, int inext) {
    // copy 48 chunks of 1 KB: chunk = kb*2+par holds B-frag (kb, nt=2*inext+par)
#pragma unroll
    for (int k = 0; k < 12; ++k) {
      int f = tid + (k << 8);          // 0..3071, 16 B per transfer
      int chunk = f >> 6;              // 0..47
      int kb = chunk >> 1, par = chunk & 1;
      unsigned int ldsOff =
          stageBase + (unsigned int)(buf * 49152 + chunk * 1024 + (f & 63) * 16);
      unsigned long long g = (unsigned long long)(uintptr_t)(
          aw1Pc + (size_t)(kb * 48 + 2 * inext + par) * 1024 + (f & 63) * 16);
      async_ld16(ldsOff, g);
    }
  };

  float acc[8] = {0.f, 0.f, 0.f, 0.f, 0.f, 0.f, 0.f, 0.f};
  issue_stage(0, 0);
  wait_async0();
  __syncthreads();
  for (int i = 0; i < 24; ++i) {
    int buf = i & 1;
    if (i + 1 < 24) issue_stage(buf ^ 1, i + 1);  // overlap copy with WMMA
    int nt = parity + i * 2;
    v8f c = {0.f, 0.f, 0.f, 0.f, 0.f, 0.f, 0.f, 0.f};
    for (int kb = 0; kb < 24; ++kb) {
      v16bf a = *(const v16bf*)(embB + ((mt * 24 + kb) * 32 + lane) * 16);
      v16bf bfr =
          *(const v16bf*)(lds + buf * 49152 + (kb * 2 + parity) * 1024 + lane * 32);
      c = wmma_bf16(a, bfr, c);
    }
    int col = nt * 16 + (lane & 15);
    float bias = ab1[col];
    float wc = aw2[col];
#pragma unroll
    for (int r = 0; r < 8; ++r) acc[r] += fmaxf(c[r] + bias, 0.f) * wc;
    wait_async0();   // each wave waits for the async copies it issued
    __syncthreads();
  }
  {
    int rowhalf = lane >> 4;
#pragma unroll
    for (int r = 0; r < 8; ++r) {
      int l = mt * 16 + rowhalf * 8 + r;
      atomicAdd(&logitsBuf[l], acc[r]);  // ds_add_f32
    }
  }
  __syncthreads();

  // Phase 4: mask + sigmoid per row l
  if (tid < 64) {
    int l = tid;
    int t1 = x1[bb * 64 + l]; if (t1 == PADID) t1 = 0;
    int t2 = x2[bb * 64 + mm]; if (t2 == PADID) t2 = 0;
    bool valid = (mask1[bb * 64 + l] != 0) && (mask2[bb * 64 + mm] != 0) && (t1 != t2);
    float lg = valid ? logitsBuf[l] : -1000000.0f;
    wBuf[l] = 1.0f / (1.0f + __expf(-lg));
  }
  __syncthreads();
  if (tid == 0) {
    float s = 0.f;
    for (int l = 0; l < 64; ++l) s += wBuf[l];
    unsafeAtomicAdd(&den[bb], s);  // global_atomic_add_f32
  }

  // Phase 5: pooled numerator: pool[b,d] += sum_l w[l]*emb[l,d]
  for (int d = tid; d < D_; d += 256) {
    int kb2 = d >> 5, kk = d & 31, khalf, jj;
    apack_inv(kk, khalf, jj);
    float s = 0.f;
    for (int l = 0; l < 64; ++l) {
      int ln2 = (khalf << 4) | (l & 15);
      int mb2 = l >> 4;
      s += wBuf[l] * (float)embB[((mb2 * 24 + kb2) * 32 + ln2) * 16 + jj];
    }
    unsafeAtomicAdd(&pool[(size_t)bb * D_ + d], s);
  }
}

// ---------------------------------------------------------------------------
// Kernel 5: y[b] = (pool[b,:] . cw) / (den[b]+1e-5) + cb
// ---------------------------------------------------------------------------
__global__ void final_kernel(const float* __restrict__ pool,
                             const float* __restrict__ den,
                             const float* __restrict__ cw,
                             const float* __restrict__ cb,
                             float* __restrict__ y) {
  __shared__ float red[256];
  int bb = blockIdx.x;
  float s = 0.f;
  for (int d = threadIdx.x; d < D_; d += 256) s += pool[(size_t)bb * D_ + d] * cw[d];
  red[threadIdx.x] = s;
  __syncthreads();
  for (int off = 128; off > 0; off >>= 1) {
    if (threadIdx.x < off) red[threadIdx.x] += red[threadIdx.x + off];
    __syncthreads();
  }
  if (threadIdx.x == 0) y[bb] = red[0] / (den[bb] + 1e-5f) + cb[0];
}

// ---------------------------------------------------------------------------
extern "C" void kernel_launch(void* const* d_in, const int* in_sizes, int n_in,
                              void* d_out, int out_size, void* d_ws, size_t ws_size,
                              hipStream_t stream) {
  const int*   x1    = (const int*)d_in[0];
  const int*   x2    = (const int*)d_in[1];
  const int*   mask1 = (const int*)d_in[2];
  const int*   mask2 = (const int*)d_in[3];
  const float* embed = (const float*)d_in[4];
  const float* tw1   = (const float*)d_in[5];
  const float* tb1   = (const float*)d_in[6];
  const float* tw2   = (const float*)d_in[7];
  const float* tb2   = (const float*)d_in[8];
  const float* aw1   = (const float*)d_in[9];
  const float* ab1   = (const float*)d_in[10];
  const float* aw2   = (const float*)d_in[11];
  const float* ab2   = (const float*)d_in[12];
  const float* cw    = (const float*)d_in[13];
  const float* cb    = (const float*)d_in[14];
  float* y = (float*)d_out;

  char* ws = (char*)d_ws;
  // workspace layout (bytes)
  bf16_t* e1A  = (bf16_t*)(ws + 0);          // 512*768*2  = 786432
  bf16_t* e2A  = (bf16_t*)(ws + 786432);     // 786432
  bf16_t* w1aP = (bf16_t*)(ws + 1572864);    // 768*1024*2 = 1572864
  bf16_t* w1bP = (bf16_t*)(ws + 3145728);    // 1572864
  bf16_t* tw2P = (bf16_t*)(ws + 4718592);    // 1024*768*2 = 1572864
  bf16_t* aw1P = (bf16_t*)(ws + 6291456);    // 768*768*2  = 1179648
  float*  h1f  = (float*)(ws + 7471104);     // 512*1024*4 = 2097152
  float*  h2f  = (float*)(ws + 9568256);     // 2097152
  float*  pool = (float*)(ws + 11665408);    // 8*768*4    = 24576
  float*  den  = (float*)(ws + 11689984);    // 8*4  (contiguous with pool)

  // 1) pack embeddings (gather) + weights to bf16 fragment layouts
  pack_embed_kernel<<<(2 * 512 * 768 + 255) / 256, 256, 0, stream>>>(x1, x2, embed,
                                                                     e1A, e2A);
  pack_B_kernel<<<(768 * 1024 + 255) / 256, 256, 0, stream>>>(tw1, w1aP, 768, 1024);
  pack_B_kernel<<<(768 * 1024 + 255) / 256, 256, 0, stream>>>(tw1 + (size_t)768 * 1024,
                                                              w1bP, 768, 1024);
  pack_B_kernel<<<(1024 * 768 + 255) / 256, 256, 0, stream>>>(tw2, tw2P, 1024, 768);
  pack_B_kernel<<<(768 * 768 + 255) / 256, 256, 0, stream>>>(aw1, aw1P, 768, 768);

  // 2) h1/h2 GEMMs (WMMA bf16, f32 out)
  gemm_h_kernel<<<256, 256, 0, stream>>>(e1A, e2A, w1aP, w1bP, h1f, h2f);

  // 3) zero accumulators (pool + den are contiguous: 6144 + 8 floats)
  zero_kernel<<<(6152 + 255) / 256, 256, 0, stream>>>(pool, 6152);

  // 4) fused main pipeline: 512 workgroups, 229888 B dynamic LDS each
  cross_main_kernel<<<512, 256, 229888, stream>>>(x1, x2, mask1, mask2, h1f, h2f, tb1,
                                                  tw2P, tb2, aw1P, ab1, aw2, ab2,
                                                  pool, den);

  // 5) finalize
  final_kernel<<<8, 256, 0, stream>>>(pool, den, cw, cb, y);
}
